// Encoder_46918222741523
// MI455X (gfx1250) — compile-verified, matrix operand order checked by reading, not run on hardware
//
#include <hip/hip_runtime.h>

// ---------------------------------------------------------------------------
// SAN encoder for MI455X (gfx1250, wave32).
//  * Big channel-mixing matmuls -> WMMA GEMM (bf16 A/B, f32 accum,
//    v_wmma_f32_16x16x32_bf16), templated on the activation leading dim so
//    all B/C accesses use immediate offsets; 2x2 tiles per wave.
//  * Tiny-K / tiny-M layers (stem K=3, stage0 q/k M=4) -> direct VALU FMA
//    kernels (WMMA would be <10% utilized; these layers are bandwidth bound).
//  * Guarded WMMA GEMM (branch-free mask-multiply) only for the FC (M=1000).
//  * BN+ReLU fused; attention unfolds are on-the-fly reflect-padded gathers.
// ---------------------------------------------------------------------------

typedef __attribute__((ext_vector_type(16))) __bf16 v16bf;
typedef __attribute__((ext_vector_type(8)))  float  v8f;

#define WMMA_BF16(a, b, c) \
  __builtin_amdgcn_wmma_f32_16x16x32_bf16(false, (a), false, (b), (short)0, (c), false, false)

__device__ __forceinline__ int refl(int t, int n) {
  t = (t < 0) ? -t : t;
  return (t >= n) ? (2 * n - 2 - t) : t;
}

// A fragment (16-bit layout, ISA 7.12.2): lane owns one row; elements 0..7 are
// K = k0 + half*8 + (0..7), elements 8..15 are K = k0 + 16 + half*8 + (0..7).
// Both runs are contiguous -> two pairs of b128 loads (immediate offsets).
__device__ __forceinline__ v16bf load_a_frag(const float* __restrict__ row,
                                             int k0, int half, bool fullk) {
  v16bf r;
  const float4* p = (const float4*)(row + k0 + half * 8);
  float4 x0 = p[0], x1 = p[1];
  r[0] = (__bf16)x0.x; r[1] = (__bf16)x0.y; r[2] = (__bf16)x0.z; r[3] = (__bf16)x0.w;
  r[4] = (__bf16)x1.x; r[5] = (__bf16)x1.y; r[6] = (__bf16)x1.z; r[7] = (__bf16)x1.w;
  if (fullk) {  // wave-uniform
    const float4* q = (const float4*)(row + k0 + 16 + half * 8);
    float4 y0 = q[0], y1 = q[1];
    r[8]  = (__bf16)y0.x; r[9]  = (__bf16)y0.y; r[10] = (__bf16)y0.z; r[11] = (__bf16)y0.w;
    r[12] = (__bf16)y1.x; r[13] = (__bf16)y1.y; r[14] = (__bf16)y1.z; r[15] = (__bf16)y1.w;
  } else {
    __bf16 z = (__bf16)0.0f;
    r[8] = z; r[9] = z; r[10] = z; r[11] = z; r[12] = z; r[13] = z; r[14] = z; r[15] = z;
  }
  return r;
}

// B fragment: lane owns one column (unit stride across lanes -> coalesced).
// LDB compile-time => one base pointer + constant offsets: the compiler can
// clause the 16 dword loads and keep a single loop-carried pointer.
template <int LDB>
__device__ __forceinline__ v16bf load_b_frag(const float* __restrict__ col,
                                             int k0, int half, bool fullk) {
  v16bf r;
  const float* base = col + (long)(k0 + half * 8) * LDB;
#pragma unroll
  for (int e = 0; e < 8; ++e) r[e] = (__bf16)base[e * LDB];
  if (fullk) {  // wave-uniform
#pragma unroll
    for (int e = 0; e < 8; ++e) r[8 + e] = (__bf16)base[(16 + e) * LDB];
  } else {
    __bf16 z = (__bf16)0.0f;
#pragma unroll
    for (int e = 0; e < 8; ++e) r[8 + e] = z;
  }
  return r;
}

// C/D layout: VGPR r -> M = mt + r + 8*half, lane -> N = nt + l16.
// LDC compile-time => stores/res reads use base + immediate offsets.
template <int LDC>
__device__ __forceinline__ void store_tile(
    float* __restrict__ Cb, v8f acc, int mt, int nt,
    const float* __restrict__ bias, const float* __restrict__ res,
    int doRelu, int half, int l16)
{
  const int mbase = mt + 8 * half;
  float* base = Cb + (long)mbase * LDC + nt + l16;
  const float* rbase = res ? res + (long)mbase * LDC + nt + l16 : nullptr;
#pragma unroll
  for (int r = 0; r < 8; ++r) {
    float v = acc[r];
    if (bias) v += bias[mbase + r];
    if (rbase) v += rbase[r * LDC];
    if (doRelu) v = fmaxf(v, 0.0f);
    base[r * LDC] = v;
  }
}

// ---------------------------------------------------------------------------
// Fast conv1x1 GEMM: C[b,m,p] = sum_k A[m,k] * B[b,k,p]  (+bias/res/relu)
// Requires: M%16==0, K%16==0, N%16==0, unit stride in p, LDB == spatial size.
// One wave owns a 32x32 macro-tile (2x2 WMMA tiles; uniform sub-tile guards
// keep EXEC all-ones as WMMA requires).
// ---------------------------------------------------------------------------
template <int LDB>
__global__ void __launch_bounds__(128) wmma_gemm_conv_kernel(
    const float* __restrict__ A, int lda,
    const float* __restrict__ B, long bsB,
    float* __restrict__ C, long bsC,
    const float* __restrict__ bias,
    const float* __restrict__ res, long bsR,
    int M, int N, int K, int doRelu)
{
  const int wave = threadIdx.x >> 5;
  const int lane = threadIdx.x & 31;
  const int half = lane >> 4, l16 = lane & 15;
  const int m0 = blockIdx.x * 32;
  const int n0 = (blockIdx.y * 4 + wave) * 32;
  if (n0 >= N) return;                       // wave-uniform
  const bool m2 = (m0 + 16) < M;             // wave-uniform sub-tile enables
  const bool n2 = (n0 + 16) < N;

  const float* Bb = B + (long)blockIdx.z * bsB;
  float*       Cb = C + (long)blockIdx.z * bsC;
  const float* Rb = res ? res + (long)blockIdx.z * bsR : nullptr;
  const float* Ar0 = A + (long)(m0 + l16) * lda;
  const float* Ar1 = A + (long)(m0 + 16 + l16) * lda;
  const float* Bc0 = Bb + (n0 + l16);
  const float* Bc1 = Bc0 + 16;

  v8f a00 = {}, a01 = {}, a10 = {}, a11 = {};
  for (int k0 = 0; k0 < K; k0 += 32) {
    const bool fullk = (k0 + 32) <= K;       // wave-uniform K tail
    if (k0 + 32 < K)
      __builtin_prefetch(Ar0 + k0 + 32, 0, 0);   // global_prefetch_b8
    v16bf af0 = load_a_frag(Ar0, k0, half, fullk);
    v16bf bf0 = load_b_frag<LDB>(Bc0, k0, half, fullk);
    a00 = WMMA_BF16(af0, bf0, a00);
    if (n2) {
      v16bf bf1 = load_b_frag<LDB>(Bc1, k0, half, fullk);
      a01 = WMMA_BF16(af0, bf1, a01);
      if (m2) {
        v16bf af1 = load_a_frag(Ar1, k0, half, fullk);
        a10 = WMMA_BF16(af1, bf0, a10);
        a11 = WMMA_BF16(af1, bf1, a11);
      }
    } else if (m2) {
      v16bf af1 = load_a_frag(Ar1, k0, half, fullk);
      a10 = WMMA_BF16(af1, bf0, a10);
    }
  }
  store_tile<LDB>(Cb, a00, m0, n0, bias, Rb, doRelu, half, l16);
  if (n2)       store_tile<LDB>(Cb, a01, m0,      n0 + 16, bias, Rb, doRelu, half, l16);
  if (m2)       store_tile<LDB>(Cb, a10, m0 + 16, n0,      bias, Rb, doRelu, half, l16);
  if (m2 && n2) store_tile<LDB>(Cb, a11, m0 + 16, n0 + 16, bias, Rb, doRelu, half, l16);
}

// ---------------------------------------------------------------------------
// Guarded GEMM for the FC layer. Branch-free: addresses are clamped in-bounds
// and the loaded value is multiplied by a 0/1 mask (the load feeds the fmul on
// every path, so the compiler cannot sink it under an EXEC branch).
// ---------------------------------------------------------------------------
__global__ void __launch_bounds__(128) wmma_gemm_kernel(
    const float* __restrict__ A, int lda,
    const float* __restrict__ B, long ldbK, long ldbN, long bsB,
    float* __restrict__ C, long ldcM, long ldcN, long bsC,
    const float* __restrict__ bias,
    const float* __restrict__ res, long bsR,
    int M, int N, int K, int doRelu)
{
  const int wave = threadIdx.x >> 5;
  const int lane = threadIdx.x & 31;
  const int m0 = (blockIdx.x * 4 + wave) * 16;
  const int n0 = blockIdx.y * 16;
  if (m0 >= M || n0 >= N) return;            // wave-uniform

  const float* Bb = B + (long)blockIdx.z * bsB;
  float*       Cb = C + (long)blockIdx.z * bsC;
  const int half = lane >> 4;
  const int l16  = lane & 15;
  const int am   = m0 + l16;
  const int bn   = n0 + l16;
  const int amc  = (am < M) ? am : (M - 1);  // clamped, always in-bounds
  const int bnc  = (bn < N) ? bn : (N - 1);
  const float amask = (am < M) ? 1.0f : 0.0f;
  const float bmask = (bn < N) ? 1.0f : 0.0f;
  const float* Arow = A + (long)amc * lda;
  const float* Bcol = Bb + (long)bnc * ldbN;

  v8f acc = {};
  for (int k0 = 0; k0 < K; k0 += 32) {
    v16bf av, bv;
#pragma unroll
    for (int e = 0; e < 16; ++e) {
      const int vv = e >> 1, lo = e & 1;
      const int k  = k0 + vv * 2 + ((vv >= 4) ? 8 : 0) + half * 8 + lo;
      const int kc = (k < K) ? k : (K - 1);
      const float kmask = (k < K) ? 1.0f : 0.0f;
      float aval = Arow[kc] * (amask * kmask);          // unconditional load
      float bval = Bcol[(long)kc * ldbK] * (bmask * kmask);
      av[e] = (__bf16)aval;
      bv[e] = (__bf16)bval;
    }
    acc = WMMA_BF16(av, bv, acc);
  }
#pragma unroll
  for (int r = 0; r < 8; ++r) {
    const int m = m0 + r + 8 * half;
    const int n = n0 + l16;
    if (m < M && n < N) {
      float v = acc[r];
      if (bias) v += bias[m];
      const long ci = (long)m * ldcM + (long)n * ldcN;
      if (res) v += res[(long)blockIdx.z * bsR + ci];
      if (doRelu) v = fmaxf(v, 0.0f);
      Cb[ci] = v;
    }
  }
}

// ---------------------------------------------------------------------------
// Stem conv1x1 (3->64) @64x64: K=3 makes WMMA <10% utilized and the layer is
// store-bandwidth bound -> direct FMA, weights via uniform scalar loads,
// hardcoded HW=4096 so the 64 output stores use immediate offsets.
// ---------------------------------------------------------------------------
__global__ void __launch_bounds__(256) stem_conv_kernel(
    const float* __restrict__ x, const float* __restrict__ w,  // w: (64,3)
    float* __restrict__ y, long total /* Bn*4096 */)
{
  constexpr int HW = 4096;
  long i = (long)blockIdx.x * blockDim.x + threadIdx.x;
  if (i >= total) return;
  int p = (int)(i % HW); int b = (int)(i / HW);
  const float* xb = x + (long)b * 3 * HW + p;
  float x0 = xb[0], x1 = xb[HW], x2 = xb[2 * HW];
  float* yb = y + (long)b * 64 * HW + p;
#pragma unroll
  for (int o = 0; o < 64; ++o)
    yb[o * HW] = w[o * 3 + 0] * x0 + w[o * 3 + 1] * x1 + w[o * 3 + 2] * x2;
}

// ---------------------------------------------------------------------------
// Stage-0 q & k projections fused (M=4 each, K=64): one pass over x computes
// both 4-channel outputs, sharing the 64 input loads. HW=1024 hardcoded.
// ---------------------------------------------------------------------------
__global__ void __launch_bounds__(256) qk_small_kernel(
    const float* __restrict__ xin,                     // (Bn,64,1024)
    const float* __restrict__ wq, const float* __restrict__ wk,  // (4,64)
    float* __restrict__ q, float* __restrict__ k,      // (Bn,4,1024)
    long total /* Bn*1024 */)
{
  constexpr int HW = 1024;
  long i = (long)blockIdx.x * blockDim.x + threadIdx.x;
  if (i >= total) return;
  int p = (int)(i % HW); int b = (int)(i / HW);
  const float* xb = xin + (long)b * 64 * HW + p;
  float aq0 = 0, aq1 = 0, aq2 = 0, aq3 = 0;
  float ak0 = 0, ak1 = 0, ak2 = 0, ak3 = 0;
#pragma unroll 8
  for (int c = 0; c < 64; ++c) {
    float xv = xb[c * HW];
    aq0 += wq[0 * 64 + c] * xv; aq1 += wq[1 * 64 + c] * xv;
    aq2 += wq[2 * 64 + c] * xv; aq3 += wq[3 * 64 + c] * xv;
    ak0 += wk[0 * 64 + c] * xv; ak1 += wk[1 * 64 + c] * xv;
    ak2 += wk[2 * 64 + c] * xv; ak3 += wk[3 * 64 + c] * xv;
  }
  float* qb = q + (long)b * 4 * HW + p;
  float* kb = k + (long)b * 4 * HW + p;
  qb[0] = aq0; qb[HW] = aq1; qb[2 * HW] = aq2; qb[3 * HW] = aq3;
  kb[0] = ak0; kb[HW] = ak1; kb[2 * HW] = ak2; kb[3 * HW] = ak3;
}

// ---------------------------------------------------------------------------
// Fused eval-mode BatchNorm (+ optional ReLU), NCHW, in-place safe.
// ---------------------------------------------------------------------------
__global__ void __launch_bounds__(256) bn_act_kernel(
    const float* __restrict__ x, float* __restrict__ y,
    const float* __restrict__ g, const float* __restrict__ b,
    const float* __restrict__ m, const float* __restrict__ v,
    int C, int HW, long total, int relu)
{
  long i = (long)blockIdx.x * blockDim.x + threadIdx.x;
  if (i >= total) return;
  int c = (int)((i / HW) % C);
  float s   = g[c] * rsqrtf(v[c] + 1e-5f);
  float val = x[i] * s + (b[c] - m[c] * s);
  y[i] = relu ? fmaxf(val, 0.0f) : val;
}

__global__ void __launch_bounds__(256) maxpool2_kernel(
    const float* __restrict__ x, float* __restrict__ y,
    int C, int Hin, int Win, int Bn)
{
  int Ho = Hin >> 1, Wo = Win >> 1;
  long total = (long)Bn * C * Ho * Wo;
  long i = (long)blockIdx.x * blockDim.x + threadIdx.x;
  if (i >= total) return;
  int wo = (int)(i % Wo); long t = i / Wo;
  int ho = (int)(t % Ho); t /= Ho;
  int c  = (int)(t % C);  int b = (int)(t / C);
  const float* p = x + (((long)(b * C + c) * Hin + ho * 2) * Win + wo * 2);
  y[i] = fmaxf(fmaxf(p[0], p[1]), fmaxf(p[Win], p[Win + 1]));
}

// ---------------------------------------------------------------------------
// Attention weight path, fused per (b, j, h, w):
//   feat = [q - gather(k), pos - gather(pos)] -> BN+ReLU -> w1 -> BN+ReLU
//   -> w2 + bias -> logits[b,g,j,h,w].   (pos bias pb cancels in subtraction)
// ---------------------------------------------------------------------------
template <int REL, int KSZ>
__global__ void __launch_bounds__(256) att_feat_kernel(
    const float* __restrict__ q, const float* __restrict__ kk,
    const float* __restrict__ pw,
    const float* __restrict__ bag, const float* __restrict__ bab,
    const float* __restrict__ bam, const float* __restrict__ bav,
    const float* __restrict__ w1,
    const float* __restrict__ bbg, const float* __restrict__ bbb,
    const float* __restrict__ bbm, const float* __restrict__ bbv,
    const float* __restrict__ w2, const float* __restrict__ w2b,
    float* __restrict__ logits, int Bn, int H, int W)
{
  constexpr int CA = REL + 2, K2 = KSZ * KSZ, PAD = KSZ / 2, G2 = REL / 2;
  const int HW = H * W;
  long total = (long)Bn * K2 * HW;
  long i = (long)blockIdx.x * blockDim.x + threadIdx.x;
  if (i >= total) return;
  int w = (int)(i % W); long t = i / W;
  int h = (int)(t % H); t /= H;
  int j = (int)(t % K2); int b = (int)(t / K2);
  int di = j / KSZ - PAD, dj = j % KSZ - PAD;
  int hn = refl(h + di, H), wn = refl(w + dj, W);

  float feat[CA];
  const long qb = (long)b * REL * HW;
#pragma unroll
  for (int c = 0; c < REL; ++c)
    feat[c] = q[qb + (long)c * HW + h * W + w] - kk[qb + (long)c * HW + hn * W + wn];
  float dlw = (float)(w - wn) * (2.0f / (float)(W - 1));
  float dlh = (float)(h - hn) * (2.0f / (float)(H - 1));
  feat[REL]     = pw[0] * dlw + pw[1] * dlh;
  feat[REL + 1] = pw[2] * dlw + pw[3] * dlh;

#pragma unroll
  for (int c = 0; c < CA; ++c) {
    float s = bag[c] * rsqrtf(bav[c] + 1e-5f);
    feat[c] = fmaxf(feat[c] * s + (bab[c] - bam[c] * s), 0.0f);
  }

  float hh[REL];
  for (int o = 0; o < REL; ++o) {
    float acc = 0.0f;
#pragma unroll
    for (int c = 0; c < CA; ++c) acc += w1[o * CA + c] * feat[c];
    float s = bbg[o] * rsqrtf(bbv[o] + 1e-5f);
    hh[o] = fmaxf(acc * s + (bbb[o] - bbm[o] * s), 0.0f);
  }

  for (int g = 0; g < G2; ++g) {
    float acc = w2b[g];
#pragma unroll
    for (int c = 0; c < REL; ++c) acc += w2[g * REL + c] * hh[c];
    logits[(((long)b * G2 + g) * K2 + j) * HW + h * W + w] = acc;
  }
}

// softmax over the k2 (neighbor) axis, in place
__global__ void __launch_bounds__(256) softmax_j_kernel(
    float* __restrict__ logits, int Bn, int G2, int k2, int HW)
{
  long total = (long)Bn * G2 * HW;
  long i = (long)blockIdx.x * blockDim.x + threadIdx.x;
  if (i >= total) return;
  int p = (int)(i % HW); long t = i / HW;
  int g = (int)(t % G2); int b = (int)(t / G2);
  float* base = logits + (((long)b * G2 + g) * k2) * HW + p;
  float mx = -3.0e38f;
  for (int j = 0; j < k2; ++j) mx = fmaxf(mx, base[(long)j * HW]);
  float s = 0.0f;
  for (int j = 0; j < k2; ++j) { float e = expf(base[(long)j * HW] - mx); base[(long)j * HW] = e; s += e; }
  float inv = 1.0f / s;
  for (int j = 0; j < k2; ++j) base[(long)j * HW] *= inv;
}

// out[b,c,h,w] = sum_j v[b,c, nbr_j(h,w)] * wgt[b, c/8, j, h, w]
__global__ void __launch_bounds__(256) att_agg_kernel(
    const float* __restrict__ v, const float* __restrict__ wgt,
    float* __restrict__ out, int Bn, int mid, int G2, int ksz, int H, int W)
{
  int k2 = ksz * ksz, pad = ksz / 2, HW = H * W;
  long total = (long)Bn * mid * HW;
  long i = (long)blockIdx.x * blockDim.x + threadIdx.x;
  if (i >= total) return;
  int w = (int)(i % W); long t = i / W;
  int h = (int)(t % H); t /= H;
  int c = (int)(t % mid); int b = (int)(t / mid);
  int g = c >> 3;  // SHARE = 8
  const float* vb = v + ((long)b * mid + c) * HW;
  const float* wb = wgt + (((long)b * G2 + g) * k2) * HW + h * W + w;
  float acc = 0.0f;
  for (int j = 0; j < k2; ++j) {
    int hn = refl(h + j / ksz - pad, H);
    int wn = refl(w + j % ksz - pad, W);
    acc += vb[hn * W + wn] * wb[(long)j * HW];
  }
  out[i] = acc;
}

__global__ void __launch_bounds__(256) meanpool_kernel(
    const float* __restrict__ x, float* __restrict__ y, int HW, long total)
{
  long i = (long)blockIdx.x * blockDim.x + threadIdx.x;
  if (i >= total) return;
  const float* p = x + i * (long)HW;
  float s = 0.0f;
  for (int t = 0; t < HW; ++t) s += p[t];
  y[i] = s / (float)HW;
}

// ---------------------------------------------------------------------------
struct BN  { const float *g, *b, *m, *v; };
struct Blk {
  BN bn1;
  const float *q, *k, *vv, *pw, *pb;
  BN bn_a; const float* w1; BN bn_b; const float *w2, *w2b;
  BN bn2; const float *cw, *cb;
};

extern "C" void kernel_launch(void* const* d_in, const int* in_sizes, int n_in,
                              void* d_out, int out_size, void* d_ws, size_t ws_size,
                              hipStream_t stream)
{
  (void)in_sizes; (void)n_in; (void)out_size; (void)ws_size;
  // -------- parameter decode: setup_inputs() insertion order, recursive -----
  int cu = 0;
  auto F = [&]() { return (const float*)d_in[cu++]; };
  const float* x = F();
  const float* conv_in = F();
  BN bn_in; bn_in.g = F(); bn_in.b = F(); bn_in.m = F(); bn_in.v = F();
  const float* stage_conv[4]; for (int s = 0; s < 4; ++s) stage_conv[s] = F();
  BN stage_bn[4];
  for (int s = 0; s < 4; ++s) { stage_bn[s].g = F(); stage_bn[s].b = F(); stage_bn[s].m = F(); stage_bn[s].v = F(); }
  Blk blk[4][2];
  for (int s = 0; s < 4; ++s)
    for (int t = 0; t < 2; ++t) {
      Blk& B = blk[s][t];
      B.bn1.g = F(); B.bn1.b = F(); B.bn1.m = F(); B.bn1.v = F();
      B.q = F(); B.k = F(); B.vv = F(); B.pw = F(); B.pb = F();
      B.bn_a.g = F(); B.bn_a.b = F(); B.bn_a.m = F(); B.bn_a.v = F();
      B.w1 = F();
      B.bn_b.g = F(); B.bn_b.b = F(); B.bn_b.m = F(); B.bn_b.v = F();
      B.w2 = F(); B.w2b = F();
      B.bn2.g = F(); B.bn2.b = F(); B.bn2.m = F(); B.bn2.v = F();
      B.cw = F(); B.cb = F();
    }
  const float* fc_w = F();
  const float* fc_b = F();

  // -------- workspace plan (floats) ----------------------------------------
  float* ws = (float*)d_ws;
  const long ACT0 = 4194304;   // (16,64,64,64) stem activation
  const long ACT1 = 1048576;   // max post-pool activation (16,64,32,32)
  float* bufA = ws;            // main activation (ping)
  float* bufS = bufA + ACT0;   // residual swap partner (pong)
  float* bufB = bufS + ACT1;   // bn1-out / pooled / attention-out temp
  float* bufQ = bufB + ACT1;   // q projection        (<= 65536)
  float* bufK = bufQ + 65536;  // k projection        (<= 65536)
  float* bufV = bufK + 65536;  // v projection        (<= 262144)
  float* bufL = bufV + 262144; // attention logits    (<= 1605632)

  const int Bn = 16;
  const int CH[4] = {64, 256, 512, 1024};
  const int KS[4] = {3, 7, 7, 7};

  // fast path: M%16==0, K%16==0, N%16==0, unit stride columns, HW templated
  auto gemm_conv = [&](const float* A, int lda, const float* Bp, int HW, long bsB,
                       float* Cp, long bsC, const float* bias,
                       const float* res, long bsR,
                       int M, int N, int K, int relu, int batch) {
    dim3 g((unsigned)((M + 31) / 32), (unsigned)((N + 127) / 128), (unsigned)batch);
    dim3 blkDim(128);
    switch (HW) {
      case 1024: wmma_gemm_conv_kernel<1024><<<g, blkDim, 0, stream>>>(
                     A, lda, Bp, bsB, Cp, bsC, bias, res, bsR, M, N, K, relu); break;
      case 256:  wmma_gemm_conv_kernel<256><<<g, blkDim, 0, stream>>>(
                     A, lda, Bp, bsB, Cp, bsC, bias, res, bsR, M, N, K, relu); break;
      case 64:   wmma_gemm_conv_kernel<64><<<g, blkDim, 0, stream>>>(
                     A, lda, Bp, bsB, Cp, bsC, bias, res, bsR, M, N, K, relu); break;
      default:   wmma_gemm_conv_kernel<16><<<g, blkDim, 0, stream>>>(
                     A, lda, Bp, bsB, Cp, bsC, bias, res, bsR, M, N, K, relu); break;
    }
  };
  auto bna = [&](const float* xi, float* yo, const BN& p, int C, int HW, int relu) {
    long total = (long)Bn * C * HW;
    bn_act_kernel<<<dim3((unsigned)((total + 255) / 256)), dim3(256), 0, stream>>>(
        xi, yo, p.g, p.b, p.m, p.v, C, HW, total, relu);
  };

  // -------- stem: direct FMA conv (3->64) @64x64, then BN + ReLU -----------
  {
    long total = (long)Bn * 4096;
    stem_conv_kernel<<<dim3((unsigned)((total + 255) / 256)), dim3(256), 0, stream>>>(
        x, conv_in, bufA, total);
    bna(bufA, bufA, bn_in, 64, 4096, 1);
  }

  float* pA = bufA;   // current activation
  float* pS = bufS;   // spare (residual destination)
  int prevC = 64, H = 64;

  for (int s = 0; s < 4; ++s) {
    const int ch = CH[s], ks = KS[s];
    H >>= 1;
    const int W = H, HW = H * W;
    const int rel = ch / 16, mid = ch / 4, G2 = mid / 8, k2 = ks * ks;

    // maxpool 2x2 : pA (prevC, 2H x 2W) -> bufB
    {
      long total = (long)Bn * prevC * HW;
      maxpool2_kernel<<<dim3((unsigned)((total + 255) / 256)), dim3(256), 0, stream>>>(
          pA, bufB, prevC, 2 * H, 2 * W, Bn);
    }
    // stage conv: prevC -> ch (fast WMMA)
    gemm_conv(stage_conv[s], prevC, bufB, HW, (long)prevC * HW,
              pA, (long)ch * HW, nullptr, nullptr, 0, ch, HW, prevC, 0, Bn);

    for (int t = 0; t < 2; ++t) {
      const Blk& bk = blk[s][t];
      // bn1 + relu -> bufB
      bna(pA, bufB, bk.bn1, ch, HW, 1);
      // q / k projections: stage0 (rel=4) -> fused direct kernel, else WMMA
      if (rel >= 16) {
        gemm_conv(bk.q, ch, bufB, HW, (long)ch * HW, bufQ, (long)rel * HW,
                  nullptr, nullptr, 0, rel, HW, ch, 0, Bn);
        gemm_conv(bk.k, ch, bufB, HW, (long)ch * HW, bufK, (long)rel * HW,
                  nullptr, nullptr, 0, rel, HW, ch, 0, Bn);
      } else {
        long total = (long)Bn * 1024;
        qk_small_kernel<<<dim3((unsigned)((total + 255) / 256)), dim3(256), 0, stream>>>(
            bufB, bk.q, bk.k, bufQ, bufK, total);
      }
      // v projection (fast WMMA; M=16 degrades to half-tile)
      gemm_conv(bk.vv, ch, bufB, HW, (long)ch * HW, bufV, (long)mid * HW,
                nullptr, nullptr, 0, mid, HW, ch, 0, Bn);
      // attention weights: feat -> MLP -> logits
      {
        long total = (long)Bn * k2 * HW;
        dim3 gF((unsigned)((total + 255) / 256));
#define ATT_ARGS bufQ, bufK, bk.pw, bk.bn_a.g, bk.bn_a.b, bk.bn_a.m, bk.bn_a.v, \
                 bk.w1, bk.bn_b.g, bk.bn_b.b, bk.bn_b.m, bk.bn_b.v,            \
                 bk.w2, bk.w2b, bufL, Bn, H, W
        switch (s) {
          case 0: att_feat_kernel<4,  3><<<gF, dim3(256), 0, stream>>>(ATT_ARGS); break;
          case 1: att_feat_kernel<16, 7><<<gF, dim3(256), 0, stream>>>(ATT_ARGS); break;
          case 2: att_feat_kernel<32, 7><<<gF, dim3(256), 0, stream>>>(ATT_ARGS); break;
          default: att_feat_kernel<64, 7><<<gF, dim3(256), 0, stream>>>(ATT_ARGS); break;
        }
#undef ATT_ARGS
      }
      {
        long total = (long)Bn * G2 * HW;
        softmax_j_kernel<<<dim3((unsigned)((total + 255) / 256)), dim3(256), 0, stream>>>(
            bufL, Bn, G2, k2, HW);
      }
      {
        long total = (long)Bn * mid * HW;
        att_agg_kernel<<<dim3((unsigned)((total + 255) / 256)), dim3(256), 0, stream>>>(
            bufV, bufL, bufB, Bn, mid, G2, ks, H, W);
      }
      // bn2 + relu (in place on bufB)
      bna(bufB, bufB, bk.bn2, mid, HW, 1);
      // cw conv (mid -> ch) + bias + residual(pA) -> pS, then swap (fast)
      gemm_conv(bk.cw, mid, bufB, HW, (long)mid * HW,
                pS, (long)ch * HW, bk.cb, pA, (long)ch * HW,
                ch, HW, mid, 0, Bn);
      float* tmp = pA; pA = pS; pS = tmp;
    }
    // stage tail BN + ReLU (in place)
    bna(pA, pA, stage_bn[s], ch, HW, 1);
    prevC = ch;
  }

  // -------- global average pool + FC ---------------------------------------
  {
    long total = (long)Bn * 1024;
    meanpool_kernel<<<dim3((unsigned)((total + 255) / 256)), dim3(256), 0, stream>>>(
        pA, bufB, 16 /*HW=4x4*/, total);
  }
  // out[b, o] = fc_w[o,:] . xm[b,:] + fc_b[o]  (M=1000 ragged -> guarded)
  {
    dim3 g((unsigned)((1000 + 63) / 64), (unsigned)((Bn + 15) / 16), 1);
    wmma_gemm_kernel<<<g, dim3(128), 0, stream>>>(
        fc_w, 1024, bufB, /*ldbK=*/1, /*ldbN=*/1024, 0,
        (float*)d_out, /*ldcM=*/1, /*ldcN=*/1000, 0,
        fc_b, nullptr, 0, 1000, Bn, 1024, 0);
  }
}
